// GraphConvolution_74062416053231
// MI455X (gfx1250) — matrix operand order repaired and checked by reading, not compiled, not run on hardware
//
#include <hip/hip_runtime.h>

typedef __attribute__((ext_vector_type(16))) _Float16     v16h;
typedef __attribute__((ext_vector_type(8)))  float        v8f;
typedef __attribute__((ext_vector_type(8)))  unsigned int v8u;
typedef unsigned int u32;

#define C_DIM   128
#define N_DIM   65536
#define B_DIM   4
#define NNB     3
#define M_TOTAL (B_DIM * N_DIM)   /* 262144 rows */
#define TILE_M  128
#define AROW    130               /* ushort stride: 260 bytes = 65 uints, kills bank conflicts */

// ---------------- K0: zero the BN stats accumulators ----------------
__global__ void zero_stats_kernel(float* __restrict__ stats) {
    stats[threadIdx.x] = 0.0f;    // 256 floats: [0..127]=sum, [128..255]=sumsq
}

// ---------------- K1: WmT[k][c] = mean_m W[c][k][m]  (f16) ----------------
__global__ __launch_bounds__(256)
void prep_wmt_kernel(const float* __restrict__ W, unsigned short* __restrict__ wmt) {
    int idx = blockIdx.x * 256 + threadIdx.x;        // 0..16383 == k*128 + c
    int k = idx >> 7;
    int c = idx & 127;
    const float* wp = W + ((size_t)c * C_DIM + k) * 16;
    float s = 0.f;
#pragma unroll
    for (int m = 0; m < 16; ++m) s += wp[m];
    s *= (1.0f / 16.0f);
    _Float16 h = (_Float16)s;
    wmt[idx] = __builtin_bit_cast(unsigned short, h);
}

// ---------------- K2: gather+mean -> f16 LDS -> WMMA GEMM -> feat + stats ----------------
__global__ __launch_bounds__(256)
void gconv_wmma_kernel(const float* __restrict__ fea,
                       const int*   __restrict__ ring,
                       const unsigned short* __restrict__ wmt,
                       float* __restrict__ feat,
                       float* __restrict__ stats) {
    __shared__ unsigned short Asm_[TILE_M * AROW];   // f_mean tile, [row][c] f16
    __shared__ unsigned short Bsm_[C_DIM * AROW];    // WmT, [k_out][c] f16
    __shared__ int   s_idx[TILE_M * 4];
    __shared__ float s_sum[C_DIM];
    __shared__ float s_sq [C_DIM];

    const int tid = threadIdx.x;
    const int r0  = blockIdx.x * TILE_M;
    const int b   = r0 >> 16;                        // N_DIM = 65536
    const int n0  = r0 & (N_DIM - 1);

    // stage WmT into LDS (uint-wide copies; row = 65 uints)
    {
        const u32* wu = (const u32*)wmt;
        u32* bs = (u32*)Bsm_;
        for (int i = tid; i < (C_DIM * C_DIM) / 2; i += 256) {
            int k  = i >> 6;
            int cu = i & 63;
            bs[k * (AROW / 2) + cu] = wu[i];
        }
    }
    // preload neighbor indices for the 128 rows
    for (int i = tid; i < TILE_M * NNB; i += 256) {
        int row = i / 3, m = i - row * 3;
        s_idx[row * 4 + m] = ring[((size_t)b * N_DIM + (size_t)(n0 + row)) * NNB + m];
    }
    if (tid < C_DIM) { s_sum[tid] = 0.f; s_sq[tid] = 0.f; }
    __syncthreads();

    // gather + average -> f16 into Asm (lanes sweep n => self term coalesced; neighbors hit L2)
    const float* feab = fea + (size_t)b * C_DIM * N_DIM;
    for (int i = tid; i < TILE_M * C_DIM; i += 256) {
        int c = i >> 7, row = i & 127;
        const float* fr = feab + (size_t)c * N_DIM;
        float v = fr[n0 + row]
                + fr[s_idx[row * 4 + 0]]
                + fr[s_idx[row * 4 + 1]]
                + fr[s_idx[row * 4 + 2]];
        v *= 0.25f;
        _Float16 h = (_Float16)v;
        Asm_[row * AROW + c] = __builtin_bit_cast(unsigned short, h);
    }
    __syncthreads();

    const int wv   = tid >> 5;       // wave 0..7 -> M strip
    const int lane = tid & 31;
    const int lm   = lane & 15;
    const int lh   = lane >> 4;      // half-wave select

    // A fragments: ISA 16-bit A 16x32 layout. lanes0-15: K pairs {0..7,16..23}; lanes16-31: {8..15,24..31}
    v16h afrag[4];
    {
        const u32* Arow = (const u32*)Asm_ + (wv * 16 + lm) * (AROW / 2);
        const int khalf = lh * 8;
#pragma unroll
        for (int ck = 0; ck < 4; ++ck) {
            v8u au;
#pragma unroll
            for (int j = 0; j < 4; ++j) {
                int k = ck * 32 + khalf + 2 * j;
                au[j]     = Arow[k >> 1];
                au[j + 4] = Arow[(k + 16) >> 1];
            }
            afrag[ck] = __builtin_bit_cast(v16h, au);
        }
    }

    const size_t featbase = (size_t)r0 * C_DIM;
#pragma unroll
    for (int nt = 0; nt < 8; ++nt) {
        // B fragment: lanes0-15 hold K {0..7,16..23} (pairs per VGPR), lanes16-31 K {8..15,24..31}; N = lane&15
        const u32* Brow = (const u32*)Bsm_ + (nt * 16 + lm) * (AROW / 2);
        const int  kb   = lh * 8;
        v8f acc = {0.f, 0.f, 0.f, 0.f, 0.f, 0.f, 0.f, 0.f};
#pragma unroll
        for (int ck = 0; ck < 4; ++ck) {
            v8u bu;
#pragma unroll
            for (int j = 0; j < 8; ++j) {
                int cc = ck * 32 + kb + 2 * (j & 3) + ((j >> 2) << 4);
                bu[j] = Brow[cc >> 1];
            }
            v16h bfrag = __builtin_bit_cast(v16h, bu);
            acc = __builtin_amdgcn_wmma_f32_16x16x32_f16(
                false, afrag[ck], false, bfrag, (short)0, acc, false, false);
        }
        // D layout: VGPR r -> M = wv*16 + lh*8 + r, N = nt*16 + (lane&15)
        const int col = nt * 16 + lm;
        float ps = 0.f, pq = 0.f;
#pragma unroll
        for (int r = 0; r < 8; ++r) {
            const int mrow = wv * 16 + lh * 8 + r;
            float v = acc[r];
            feat[featbase + (size_t)mrow * C_DIM + col] = v;
            ps += v;
            pq += v * v;
        }
        atomicAdd(&s_sum[col], ps);
        atomicAdd(&s_sq [col], pq);
    }
    __syncthreads();
    if (tid < C_DIM) {
        atomicAdd(&stats[tid],          s_sum[tid]);
        atomicAdd(&stats[C_DIM + tid],  s_sq [tid]);
    }
}

// ---------------- K3: BN + ReLU + transpose to [B, K, N] ----------------
__global__ __launch_bounds__(256)
void bn_relu_transpose_kernel(const float* __restrict__ feat,
                              const float* __restrict__ stats,
                              const float* __restrict__ gamma,
                              const float* __restrict__ beta,
                              float* __restrict__ out) {
    __shared__ float tile[C_DIM][33];
    __shared__ float s_scale[C_DIM], s_bias[C_DIM];
    const int tid = threadIdx.x;
    if (tid < C_DIM) {
        const float invM = 1.0f / (float)M_TOTAL;
        float mu  = stats[tid] * invM;
        float var = stats[C_DIM + tid] * invM - mu * mu;
        float sc  = gamma[tid] * rsqrtf(var + 1e-5f);
        s_scale[tid] = sc;
        s_bias[tid]  = beta[tid] - mu * sc;
    }
    const size_t r0 = (size_t)blockIdx.x * 32;
    for (int i = tid; i < 32 * C_DIM; i += 256) {     // coalesced feat read
        int row = i >> 7, k = i & 127;
        tile[k][row] = feat[(r0 + row) * C_DIM + k];
    }
    __syncthreads();
    const int b  = (int)(r0 >> 16);
    const int n0 = (int)(r0 & (N_DIM - 1));
    for (int i = tid; i < 32 * C_DIM; i += 256) {     // coalesced out write
        int k = i >> 5, nl = i & 31;
        float v = tile[k][nl] * s_scale[k] + s_bias[k];
        out[((size_t)b * C_DIM + k) * N_DIM + (size_t)(n0 + nl)] = fmaxf(v, 0.f);
    }
}

// ---------------- launch ----------------
extern "C" void kernel_launch(void* const* d_in, const int* in_sizes, int n_in,
                              void* d_out, int out_size, void* d_ws, size_t ws_size,
                              hipStream_t stream) {
    const float* fea   = (const float*)d_in[0];
    const int*   ring  = (const int*)d_in[1];
    const float* W     = (const float*)d_in[2];
    const float* gamma = (const float*)d_in[3];
    const float* beta  = (const float*)d_in[4];
    float* out = (float*)d_out;

    char* ws = (char*)d_ws;
    float*          feat  = (float*)ws;                                          // 134,217,728 B
    unsigned short* wmt   = (unsigned short*)(ws + (size_t)M_TOTAL * C_DIM * 4); // 32,768 B
    float*          stats = (float*)(ws + (size_t)M_TOTAL * C_DIM * 4 + (size_t)C_DIM * C_DIM * 2);

    zero_stats_kernel<<<1, 256, 0, stream>>>(stats);
    prep_wmt_kernel<<<(C_DIM * C_DIM) / 256, 256, 0, stream>>>(W, wmt);
    gconv_wmma_kernel<<<M_TOTAL / TILE_M, 256, 0, stream>>>(fea, ring, wmt, feat, stats);
    bn_relu_transpose_kernel<<<M_TOTAL / 32, 256, 0, stream>>>(feat, stats, gamma, beta, out);
}
